// TemporalAttention_1632087573253
// MI455X (gfx1250) — compile-verified, matrix operand order checked by reading, not run on hardware
//
#include <hip/hip_runtime.h>

typedef __bf16 bf16;
typedef __attribute__((ext_vector_type(16))) __bf16 v16bf;
typedef __attribute__((ext_vector_type(8)))  float  v8f;
typedef __attribute__((ext_vector_type(4)))  int    v4i;

#define SEQ    4096
#define DMODEL 512
#define HEADS  8
#define HDIM   64
#define BATCH  2

#define AS1 __attribute__((address_space(1)))
#define AS3 __attribute__((address_space(3)))

union ABfrag { v16bf v; unsigned u[8]; };
union Pair   { unsigned u; bf16 h[2]; };

__device__ __forceinline__ v8f wmma_bf16(const ABfrag& a, const ABfrag& b, v8f c) {
  return __builtin_amdgcn_wmma_f32_16x16x32_bf16(false, a.v, false, b.v, (short)0, c, false, false);
}

// async global->LDS 16B copy (CDNA5 GLOBAL_LOAD_ASYNC_TO_LDS_B128, ASYNCcnt)
__device__ __forceinline__ void async_ld_b128(void* lds, const void* g) {
#if __has_builtin(__builtin_amdgcn_global_load_async_to_lds_b128)
  __builtin_amdgcn_global_load_async_to_lds_b128((AS1 v4i*)(g), (AS3 v4i*)(lds), 0, 0);
#else
  *reinterpret_cast<uint4*>(lds) = *reinterpret_cast<const uint4*>(g);
#endif
}

__device__ __forceinline__ void wait_async0() {
#if __has_builtin(__builtin_amdgcn_s_wait_asynccnt)
  __builtin_amdgcn_s_wait_asynccnt(0);
#else
  asm volatile("s_wait_asynccnt 0" ::: "memory");
#endif
}

// ---------------------------------------------------------------- fp32->bf16
__global__ __launch_bounds__(256)
void cvt_f32_bf16(const float* __restrict__ src, bf16* __restrict__ dst, int n) {
  int i = blockIdx.x * 256 + threadIdx.x;
  if (i < n) dst[i] = (bf16)src[i];
}

// ------------------------------------------------- tiled WMMA GEMM (bf16 in)
// C[M,N] = A[M,K] * W[K,N] + bias ; block tile 128x128, 8 waves of 32x64
template<bool OUT_F32>
__global__ __launch_bounds__(256)
void gemm_bf16_wmma(const bf16* __restrict__ A, const bf16* __restrict__ W,
                    const float* __restrict__ bias, void* __restrict__ Cout,
                    int M, int N, int K)
{
  __shared__ bf16 Asm[128 * 40];   // [row][k] padded (80B row stride, 16B aligned)
  __shared__ bf16 Bsm[128 * 40];   // [n][k]  (W transposed) padded
  const int tid  = threadIdx.x;
  const int wid  = tid >> 5, lane = tid & 31;
  const int ln   = lane & 15, half = lane >> 4;
  const int wr   = wid >> 1, wc = wid & 1;          // 4x2 wave grid
  const int rowBase = blockIdx.x * 128;
  const int colBase = blockIdx.y * 128;

  const v8f vzero = {0.f,0.f,0.f,0.f,0.f,0.f,0.f,0.f};
  v8f acc[2][4];
  #pragma unroll
  for (int mt = 0; mt < 2; ++mt)
    #pragma unroll
    for (int nt = 0; nt < 4; ++nt) acc[mt][nt] = vzero;

  for (int k0 = 0; k0 < K; k0 += 32) {
    // A tile: 128 rows x 64B, async global->LDS b128 (2 per thread)
    #pragma unroll
    for (int i = 0; i < 2; ++i) {
      int p = tid + i * 256;            // 0..511
      int r = p >> 2, q = p & 3;        // 4 x 16B per row
      async_ld_b128(&Asm[r * 40 + q * 8],
                    A + (size_t)(rowBase + r) * K + k0 + q * 8);
    }
    // prefetch next W tile (64 x 128B cachelines)
    if (k0 + 32 < K && tid < 64) {
      int kk = tid >> 1, co = (tid & 1) * 64;
      __builtin_prefetch(W + (size_t)(k0 + 32 + kk) * N + colBase + co, 0, 1);
    }
    // W tile transposed into [n][k] (manual: transpose in flight)
    #pragma unroll
    for (int i = 0; i < 8; ++i) {
      int p = tid + i * 256;
      int kk = p >> 6, np = p & 63;
      Pair c; c.u = *(const unsigned*)(W + (size_t)(k0 + kk) * N + colBase + np * 2);
      Bsm[(np * 2)     * 40 + kk] = c.h[0];
      Bsm[(np * 2 + 1) * 40 + kk] = c.h[1];
    }
    wait_async0();
    __syncthreads();

    ABfrag af[2], bfr[4];
    #pragma unroll
    for (int mt = 0; mt < 2; ++mt) {
      int r = wr * 32 + mt * 16 + ln;
      #pragma unroll
      for (int v = 0; v < 8; ++v) {
        int k = ((v & 4) << 2) + half * 8 + (v & 3) * 2;
        af[mt].u[v] = *(const unsigned*)(&Asm[r * 40 + k]);
      }
    }
    #pragma unroll
    for (int nt = 0; nt < 4; ++nt) {
      int r = wc * 64 + nt * 16 + ln;
      #pragma unroll
      for (int v = 0; v < 8; ++v) {
        int k = ((v & 4) << 2) + half * 8 + (v & 3) * 2;
        bfr[nt].u[v] = *(const unsigned*)(&Bsm[r * 40 + k]);
      }
    }
    #pragma unroll
    for (int mt = 0; mt < 2; ++mt)
      #pragma unroll
      for (int nt = 0; nt < 4; ++nt)
        acc[mt][nt] = wmma_bf16(af[mt], bfr[nt], acc[mt][nt]);
    __syncthreads();
  }

  #pragma unroll
  for (int mt = 0; mt < 2; ++mt)
    #pragma unroll
    for (int nt = 0; nt < 4; ++nt) {
      int col = colBase + wc * 64 + nt * 16 + ln;
      float bv = bias[col];
      #pragma unroll
      for (int r = 0; r < 8; ++r) {
        int row = rowBase + wr * 32 + mt * 16 + r + 8 * half;
        float val = acc[mt][nt][r] + bv;
        if (OUT_F32) ((float*)Cout)[(size_t)row * N + col] = val;
        else         ((bf16*)Cout)[(size_t)row * N + col] = (bf16)val;
      }
    }
}

// -------------------------------------------------------------------- RoPE
__global__ __launch_bounds__(256)
void rope_kernel(const bf16* __restrict__ qi, const bf16* __restrict__ ki,
                 bf16* __restrict__ qo, bf16* __restrict__ ko, int n)
{
  int idx = blockIdx.x * 256 + threadIdx.x;
  if (idx >= n) return;
  int col = idx & (DMODEL - 1);
  int row = idx >> 9;                     // b*S + s
  int s   = row & (SEQ - 1);
  int d   = col & (HDIM - 1);
  int j   = d & (HDIM / 2 - 1);
  // inv_freq = 10000^(-j/32) = exp(-j * ln(10000)/32)
  float inv = __expf(-0.28782313662425573f * (float)j);
  float sn, cs;
  __sincosf((float)s * inv, &sn, &cs);
  int   pidx = (d < HDIM / 2) ? idx + HDIM / 2 : idx - HDIM / 2;
  float sgn  = (d < HDIM / 2) ? -1.f : 1.f;
  float qv = (float)qi[idx], qp = (float)qi[pidx];
  qo[idx] = (bf16)(qv * cs + sgn * qp * sn);
  float kv = (float)ki[idx], kp = (float)ki[pidx];
  ko[idx] = (bf16)(kv * cs + sgn * kp * sn);
}

// --------------------------------------------------------- flash attention
// grid.x = S/128 query tiles, grid.y = B*H ; 8 waves, 16 query rows per wave
__global__ __launch_bounds__(256)
void flash_attn_wmma(const bf16* __restrict__ Q, const bf16* __restrict__ K,
                     const bf16* __restrict__ V, bf16* __restrict__ O)
{
  __shared__ bf16 Kt[64 * 72];        // [key][d]   (B-frag for Q*K^T), 144B rows
  __shared__ bf16 Vt[64 * 72];        // [d][key]   (B-frag for P*V)
  __shared__ bf16 Pst[8][16 * 72];    // per-wave P staging [m][key]

  const int tid  = threadIdx.x;
  const int wid  = tid >> 5, lane = tid & 31;
  const int ln   = lane & 15, half = lane >> 4;
  const int bh   = blockIdx.y;
  const int b    = bh >> 3, h = bh & 7;
  const size_t rowBase = (size_t)b * SEQ;
  const int hoff = h * HDIM;
  const int q0   = blockIdx.x * 128;

  // load Q fragments once (A layout, two 16x32 chunks over hd=64)
  ABfrag qf[2];
  {
    const bf16* qp = Q + (rowBase + q0 + wid * 16 + ln) * DMODEL + hoff;
    #pragma unroll
    for (int f = 0; f < 2; ++f)
      #pragma unroll
      for (int v = 0; v < 8; ++v) {
        int d = f * 32 + ((v & 4) << 2) + half * 8 + (v & 3) * 2;
        qf[f].u[v] = *(const unsigned*)(qp + d);
      }
  }

  const v8f vzero = {0.f,0.f,0.f,0.f,0.f,0.f,0.f,0.f};
  v8f o_[4];
  #pragma unroll
  for (int nt = 0; nt < 4; ++nt) o_[nt] = vzero;
  float m_i[8], l_i[8];
  #pragma unroll
  for (int r = 0; r < 8; ++r) { m_i[r] = -1e30f; l_i[r] = 0.f; }

  for (int kt = 0; kt < SEQ / 64; ++kt) {
    int key0 = kt * 64;
    // K tile: straight copy, async global->LDS b128 (2 per thread)
    #pragma unroll
    for (int i = 0; i < 2; ++i) {
      int p = tid + i * 256;          // 0..511
      int key = p >> 3, q = p & 7;    // 8 x 16B per 128B key row
      async_ld_b128(&Kt[key * 72 + q * 8],
                    K + (rowBase + key0 + key) * DMODEL + hoff + q * 8);
    }
    // V tile: transposed into [d][key] (manual)
    #pragma unroll
    for (int i = 0; i < 8; ++i) {
      int p = tid + i * 256;
      int key = p >> 5, dp = p & 31;
      Pair c; c.u = *(const unsigned*)(V + (rowBase + key0 + key) * DMODEL + hoff + dp * 2);
      Vt[(dp * 2)     * 72 + key] = c.h[0];
      Vt[(dp * 2 + 1) * 72 + key] = c.h[1];
    }
    wait_async0();
    __syncthreads();

    // S = Q * K^T  (16 q-rows x 64 keys per wave)
    v8f s[4];
    #pragma unroll
    for (int nt = 0; nt < 4; ++nt) s[nt] = vzero;
    #pragma unroll
    for (int nt = 0; nt < 4; ++nt)
      #pragma unroll
      for (int c = 0; c < 2; ++c) {
        ABfrag kf;
        #pragma unroll
        for (int v = 0; v < 8; ++v) {
          int d = c * 32 + ((v & 4) << 2) + half * 8 + (v & 3) * 2;
          kf.u[v] = *(const unsigned*)(&Kt[(nt * 16 + ln) * 72 + d]);
        }
        s[nt] = wmma_bf16(qf[c], kf, s[nt]);
      }

    // online softmax (rows live in 16-lane halves of the C layout)
    const float scale = 0.125f;   // hd^-0.5
    float alpha[8];
    #pragma unroll
    for (int r = 0; r < 8; ++r) {
      float v0 = s[0][r] * scale, v1 = s[1][r] * scale;
      float v2 = s[2][r] * scale, v3 = s[3][r] * scale;
      float mx = fmaxf(fmaxf(v0, v1), fmaxf(v2, v3));
      #pragma unroll
      for (int off = 1; off < 16; off <<= 1)
        mx = fmaxf(mx, __shfl_xor(mx, off, 32));
      float mn = fmaxf(m_i[r], mx);
      float p0 = __expf(v0 - mn), p1 = __expf(v1 - mn);
      float p2 = __expf(v2 - mn), p3 = __expf(v3 - mn);
      s[0][r] = p0; s[1][r] = p1; s[2][r] = p2; s[3][r] = p3;
      float rs = p0 + p1 + p2 + p3;
      #pragma unroll
      for (int off = 1; off < 16; off <<= 1)
        rs += __shfl_xor(rs, off, 32);
      alpha[r] = __expf(m_i[r] - mn);
      l_i[r]   = l_i[r] * alpha[r] + rs;
      m_i[r]   = mn;
    }
    #pragma unroll
    for (int nt = 0; nt < 4; ++nt)
      #pragma unroll
      for (int r = 0; r < 8; ++r) o_[nt][r] *= alpha[r];

    // stage P (C layout -> LDS row-major) then reload as A fragments
    #pragma unroll
    for (int nt = 0; nt < 4; ++nt)
      #pragma unroll
      for (int r = 0; r < 8; ++r)
        Pst[wid][(r + 8 * half) * 72 + nt * 16 + ln] = (bf16)s[nt][r];

    ABfrag pf[2];
    #pragma unroll
    for (int f = 0; f < 2; ++f)
      #pragma unroll
      for (int v = 0; v < 8; ++v) {
        int kk = f * 32 + ((v & 4) << 2) + half * 8 + (v & 3) * 2;
        pf[f].u[v] = *(const unsigned*)(&Pst[wid][ln * 72 + kk]);
      }

    // O += P * V
    #pragma unroll
    for (int nt = 0; nt < 4; ++nt)
      #pragma unroll
      for (int c = 0; c < 2; ++c) {
        ABfrag vf;
        #pragma unroll
        for (int v = 0; v < 8; ++v) {
          int kk = c * 32 + ((v & 4) << 2) + half * 8 + (v & 3) * 2;
          vf.u[v] = *(const unsigned*)(&Vt[(nt * 16 + ln) * 72 + kk]);
        }
        o_[nt] = wmma_bf16(pf[c], vf, o_[nt]);
      }
    __syncthreads();
  }

  #pragma unroll
  for (int nt = 0; nt < 4; ++nt)
    #pragma unroll
    for (int r = 0; r < 8; ++r) {
      float rinv = 1.0f / l_i[r];
      size_t row = rowBase + q0 + wid * 16 + r + 8 * half;
      O[row * DMODEL + hoff + nt * 16 + ln] = (bf16)(o_[nt][r] * rinv);
    }
}

// ---------------------------------------------------------------- launcher
extern "C" void kernel_launch(void* const* d_in, const int* in_sizes, int n_in,
                              void* d_out, int out_size, void* d_ws, size_t ws_size,
                              hipStream_t stream)
{
  (void)in_sizes; (void)n_in; (void)out_size; (void)ws_size;
  const float* x  = (const float*)d_in[0];
  const float* Wq = (const float*)d_in[1];
  const float* bq = (const float*)d_in[2];
  const float* Wk = (const float*)d_in[3];
  const float* bk = (const float*)d_in[4];
  const float* Wv = (const float*)d_in[5];
  const float* bv = (const float*)d_in[6];
  const float* Wo = (const float*)d_in[7];
  const float* bo = (const float*)d_in[8];

  const size_t BS = (size_t)BATCH * SEQ;        // 8192 rows
  const size_t nX = BS * DMODEL;                // 4,194,304
  const size_t nW = (size_t)DMODEL * DMODEL;    // 262,144

  char* ws = (char*)d_ws;
  size_t off = 0;
  auto take = [&](size_t elems) { bf16* p = (bf16*)(ws + off); off += elems * sizeof(bf16); return p; };
  bf16* xb   = take(nX);
  bf16* wqb  = take(nW);
  bf16* wkb  = take(nW);
  bf16* wvb  = take(nW);
  bf16* wob  = take(nW);
  bf16* qraw = take(nX);
  bf16* kraw = take(nX);
  bf16* vraw = take(nX);
  bf16* qr   = take(nX);
  bf16* kr   = take(nX);
  bf16* attn = take(nX);

  dim3 blk(256);
  // 1) fp32 -> bf16
  cvt_f32_bf16<<<dim3((unsigned)((nX + 255) / 256)), blk, 0, stream>>>(x,  xb,  (int)nX);
  cvt_f32_bf16<<<dim3((unsigned)((nW + 255) / 256)), blk, 0, stream>>>(Wq, wqb, (int)nW);
  cvt_f32_bf16<<<dim3((unsigned)((nW + 255) / 256)), blk, 0, stream>>>(Wk, wkb, (int)nW);
  cvt_f32_bf16<<<dim3((unsigned)((nW + 255) / 256)), blk, 0, stream>>>(Wv, wvb, (int)nW);
  cvt_f32_bf16<<<dim3((unsigned)((nW + 255) / 256)), blk, 0, stream>>>(Wo, wob, (int)nW);

  // 2) Q/K/V projections (WMMA)
  dim3 ggrid((unsigned)(BS / 128), DMODEL / 128);  // 64 x 4
  gemm_bf16_wmma<false><<<ggrid, blk, 0, stream>>>(xb, wqb, bq, qraw, (int)BS, DMODEL, DMODEL);
  gemm_bf16_wmma<false><<<ggrid, blk, 0, stream>>>(xb, wkb, bk, kraw, (int)BS, DMODEL, DMODEL);
  gemm_bf16_wmma<false><<<ggrid, blk, 0, stream>>>(xb, wvb, bv, vraw, (int)BS, DMODEL, DMODEL);

  // 3) RoPE on Q,K
  rope_kernel<<<dim3((unsigned)((nX + 255) / 256)), blk, 0, stream>>>(qraw, kraw, qr, kr, (int)nX);

  // 4) flash attention (WMMA)
  dim3 agrid(SEQ / 128, BATCH * HEADS);            // 32 x 16
  flash_attn_wmma<<<agrid, blk, 0, stream>>>(qr, kr, vraw, attn);

  // 5) output projection (WMMA, fp32 out)
  gemm_bf16_wmma<true><<<ggrid, blk, 0, stream>>>(attn, wob, bo, d_out, (int)BS, DMODEL, DMODEL);
}
